// Net_one_82660940578899
// MI455X (gfx1250) — matrix operand order repaired and checked by reading, not compiled, not run on hardware
//
#include <hip/hip_runtime.h>

typedef __attribute__((ext_vector_type(16))) _Float16 v16h;
typedef __attribute__((ext_vector_type(8)))  _Float16 v8h;
typedef __attribute__((ext_vector_type(4)))  _Float16 v4h;
typedef __attribute__((ext_vector_type(8)))  float    v8f;

// LDS-typed pointers (raw 32-bit LDS byte offsets -> ds_load/ds_store)
typedef __attribute__((address_space(3))) const v8h*  lds_v8h_t;
typedef __attribute__((address_space(3))) const v16h* lds_v16h_t;

#define B_DIM 8192
#define D_DIM 2048
#define H_DIM 2048
#define C_DIM 1000
#define C_PAD 1024

// LDS layout (byte offsets inside the kernel's single __shared__ block,
// assumed to start at LDS offset 0). Row stride 80B: 128x32 f16 tile with
// 16B pad per row -> fragment reads are bank-conflict-free (stride 20 dwords).
#define LDS_ROWSTR 80u
#define LDS_A0     0u
#define LDS_A1     10240u
#define LDS_B0     20480u
#define LDS_B1     30720u
#define LDS_RED    40960u
#define LDS_TOTAL  41984

// ---------------------------------------------------------------------------
// Async staging: copy one 128x32 f16 tile (rows rowblk..rowblk+127 of a
// row-major [*, ld] matrix, K-window [k0, k0+32)) into LDS at lds_base.
// 512 16-byte chunks, 2 per thread, via CDNA5 global_load_async_to_lds_b128
// (tracked by ASYNCcnt).
// ---------------------------------------------------------------------------
__device__ __forceinline__ void stage_tile_async(const _Float16* __restrict__ g,
                                                 int ld, int rowblk, int k0,
                                                 unsigned lds_base) {
  const int tid = threadIdx.x;
#pragma unroll
  for (int j = 0; j < 2; ++j) {
    const int c   = tid + j * 256;   // 0..511
    const int row = c >> 2;          // 0..127
    const int col = c & 3;           // 16B chunk within 64B row
    const void* gp = g + (size_t)(rowblk + row) * ld + k0 + col * 8;
    const unsigned lp = lds_base + (unsigned)row * LDS_ROWSTR + (unsigned)col * 16u;
    asm volatile("global_load_async_to_lds_b128 %0, %1, off"
                 :: "v"(lp), "v"(gp)
                 : "memory");
  }
}

__device__ __forceinline__ void wait_async_and_barrier() {
  asm volatile("s_wait_asynccnt 0x0" ::: "memory");
  __syncthreads();
}

// ---------------------------------------------------------------------------
// WMMA fragments from LDS (this code's contiguous K order, same for A and B, wave32)
// A 16x32 f16: lane L -> row M=L%16; halves e=0..15 -> K = h*16 + e
// B 32x16 f16: lane L -> col N=L%16; halves e=0..15 -> K = h*16 + e
// ---------------------------------------------------------------------------
__device__ __forceinline__ v16h lds_a_frag(unsigned buf, int wrow, int hsel) {
  const unsigned off = buf + (unsigned)wrow * LDS_ROWSTR + (unsigned)hsel * 32u;
  v8h lo = *(lds_v8h_t)off;
  v8h hi = *(lds_v8h_t)(off + 16u);
  return __builtin_shufflevector(lo, hi, 0,1,2,3,4,5,6,7,8,9,10,11,12,13,14,15);
}

__device__ __forceinline__ v16h lds_b_frag(unsigned buf, int wrow, int hsel) {
  const unsigned off = buf + (unsigned)wrow * LDS_ROWSTR + (unsigned)hsel * 32u;
  return *(lds_v16h_t)off;
}

// ---------------------------------------------------------------------------
// Shared GEMM core: acc[4][2] += A[128 rows @ mblk] x B[128 rows @ nblk]^T
// over Kdim, double-buffered async LDS staging. Block=128x128, 8 waves,
// wave = 64(M) x 32(N) = 4x2 WMMA tiles.
// ---------------------------------------------------------------------------
__device__ __forceinline__ void gemm_core(const _Float16* __restrict__ Ag,
                                          const _Float16* __restrict__ Bg,
                                          int ld, int Kdim, int mblk, int nblk,
                                          v8f (&acc)[4][2]) {
  const int tid  = threadIdx.x;
  const int lane = tid & 31;
  const int wave = tid >> 5;
  const int wm = wave & 1;   // 0..1
  const int wn = wave >> 1;  // 0..3
  const int hsel = lane >> 4;
  const int lrow = lane & 15;

  // prologue: stage k0=0 into buffer 0
  stage_tile_async(Ag, ld, mblk, 0, LDS_A0);
  stage_tile_async(Bg, ld, nblk, 0, LDS_B0);
  wait_async_and_barrier();

  for (int k0 = 0; k0 < Kdim; k0 += 32) {
    const int db = (k0 >> 5) & 1;
    if (k0 + 32 < Kdim) {  // stage next K-step into the other buffer
      stage_tile_async(Ag, ld, mblk, k0 + 32, db ? LDS_A0 : LDS_A1);
      stage_tile_async(Bg, ld, nblk, k0 + 32, db ? LDS_B0 : LDS_B1);
    }
    const unsigned abuf = db ? LDS_A1 : LDS_A0;
    const unsigned bbuf = db ? LDS_B1 : LDS_B0;

    v16h a[4], b[2];
#pragma unroll
    for (int i = 0; i < 4; ++i)
      a[i] = lds_a_frag(abuf, wm * 64 + i * 16 + lrow, hsel);
#pragma unroll
    for (int j = 0; j < 2; ++j)
      b[j] = lds_b_frag(bbuf, wn * 32 + j * 16 + lrow, hsel);
#pragma unroll
    for (int i = 0; i < 4; ++i)
#pragma unroll
      for (int j = 0; j < 2; ++j)
        acc[i][j] = __builtin_amdgcn_wmma_f32_16x16x32_f16(
            false, a[i], false, b[j], (short)0, acc[i][j], false, false);

    wait_async_and_barrier();
  }
}

// ---------------------------------------------------------------------------
// GEMM1: acc = x_f16 [B,D] @ signW1_f16 [H,D]^T
// epilogue: hp = scale_w*acc + b1[n]; mask=(hp>0) f16 {0,1};
//           per-block sum |hp| -> partials (deterministic LDS tree)
// ---------------------------------------------------------------------------
__global__ __launch_bounds__(256) void k_gemm1(
    const _Float16* __restrict__ xh,
    const _Float16* __restrict__ wsh,
    const float* __restrict__ b1,
    const float* __restrict__ scale_w_p,
    _Float16* __restrict__ maskh,
    float* __restrict__ partials) {
  __shared__ char lds_raw[LDS_TOTAL];
  float* red = (float*)(lds_raw + LDS_RED);  // keeps lds_raw allocated

  const int tid  = threadIdx.x;
  const int lane = tid & 31;
  const int wave = tid >> 5;
  const int mbase = blockIdx.y * 128 + (wave & 1) * 64;
  const int nbase = blockIdx.x * 128 + (wave >> 1) * 32;
  const int hsel = lane >> 4;
  const int lrow = lane & 15;

  v8f zero = {};
  v8f acc[4][2];
#pragma unroll
  for (int i = 0; i < 4; ++i)
#pragma unroll
    for (int j = 0; j < 2; ++j) acc[i][j] = zero;

  gemm_core(xh, wsh, D_DIM, D_DIM, blockIdx.y * 128, blockIdx.x * 128, acc);

  const float sw = *scale_w_p;
  float local = 0.0f;
#pragma unroll
  for (int i = 0; i < 4; ++i) {
#pragma unroll
    for (int j = 0; j < 2; ++j) {
      const int n = nbase + j * 16 + lrow;
      const float bias = b1[n];
#pragma unroll
      for (int r = 0; r < 8; ++r) {
        const int m = mbase + i * 16 + r + hsel * 8;
        const float hp = sw * acc[i][j][r] + bias;
        local += fabsf(hp);
        maskh[(size_t)m * H_DIM + n] = hp > 0.0f ? (_Float16)1.0f : (_Float16)0.0f;
      }
    }
  }

  red[tid] = local;
  __syncthreads();
  for (int s = 128; s > 0; s >>= 1) {
    if (tid < s) red[tid] += red[tid + s];
    __syncthreads();
  }
  if (tid == 0) partials[blockIdx.y * gridDim.x + blockIdx.x] = red[0];
}

// ---------------------------------------------------------------------------
// GEMM2: out = scale_h * (mask_f16 [B,H] @ W2_f16 [C_PAD,H]^T) + b2
// ---------------------------------------------------------------------------
__global__ __launch_bounds__(256) void k_gemm2(
    const _Float16* __restrict__ mh,
    const _Float16* __restrict__ w2h,
    const float* __restrict__ b2,
    const float* __restrict__ scale_h_p,
    float* __restrict__ out) {
  __shared__ char lds_raw[LDS_TOTAL];
  const int tid = threadIdx.x;
  if (tid == 0) *((volatile char*)lds_raw + LDS_RED) = 0;  // keep lds_raw live

  const int lane = tid & 31;
  const int wave = tid >> 5;
  const int mbase = blockIdx.y * 128 + (wave & 1) * 64;
  const int nbase = blockIdx.x * 128 + (wave >> 1) * 32;
  const int hsel = lane >> 4;
  const int lrow = lane & 15;

  v8f zero = {};
  v8f acc[4][2];
#pragma unroll
  for (int i = 0; i < 4; ++i)
#pragma unroll
    for (int j = 0; j < 2; ++j) acc[i][j] = zero;

  gemm_core(mh, w2h, H_DIM, H_DIM, blockIdx.y * 128, blockIdx.x * 128, acc);

  const float sh = *scale_h_p;
#pragma unroll
  for (int i = 0; i < 4; ++i) {
#pragma unroll
    for (int j = 0; j < 2; ++j) {
      const int n = nbase + j * 16 + lrow;
      if (n < C_DIM) {
        const float bias = b2[n];
#pragma unroll
        for (int r = 0; r < 8; ++r) {
          const int m = mbase + i * 16 + r + hsel * 8;
          out[(size_t)m * C_DIM + n] = sh * acc[i][j][r] + bias;
        }
      }
    }
  }
}

// ---------------------------------------------------------------------------
// Reductions (deterministic: fixed grid, LDS tree; no float atomics)
// ---------------------------------------------------------------------------
__global__ __launch_bounds__(256) void k_reduce_abs(const float* __restrict__ src,
                                                    long long n,
                                                    float* __restrict__ partials) {
  long long i = (long long)blockIdx.x * 256 + threadIdx.x;
  const long long stride = (long long)gridDim.x * 256;
  float s = 0.0f;
  for (; i < n; i += stride) s += fabsf(src[i]);
  __shared__ float red[256];
  red[threadIdx.x] = s;
  __syncthreads();
  for (int t = 128; t > 0; t >>= 1) {
    if (threadIdx.x < t) red[threadIdx.x] += red[threadIdx.x + t];
    __syncthreads();
  }
  if (threadIdx.x == 0) partials[blockIdx.x] = red[0];
}

__global__ __launch_bounds__(256) void k_finalize(const float* __restrict__ partials,
                                                  int n, float scale,
                                                  float* __restrict__ outp) {
  __shared__ float red[256];
  float s = 0.0f;
  for (int i = threadIdx.x; i < n; i += 256) s += partials[i];
  red[threadIdx.x] = s;
  __syncthreads();
  for (int t = 128; t > 0; t >>= 1) {
    if (threadIdx.x < t) red[threadIdx.x] += red[threadIdx.x + t];
    __syncthreads();
  }
  if (threadIdx.x == 0) *outp = red[0] * scale;
}

// ---------------------------------------------------------------------------
// Elementwise converters (4 elems/thread, exact grid coverage)
// ---------------------------------------------------------------------------
__global__ __launch_bounds__(256) void k_cvt_f16(const float* __restrict__ src,
                                                 _Float16* __restrict__ dst,
                                                 long long n) {
  long long i = ((long long)blockIdx.x * 256 + threadIdx.x) * 4;
  if (i + 3 < n) {
    float4 v = *(const float4*)(src + i);
    v4h o;
    o[0] = (_Float16)v.x; o[1] = (_Float16)v.y;
    o[2] = (_Float16)v.z; o[3] = (_Float16)v.w;
    *(v4h*)(dst + i) = o;
  }
}

__global__ __launch_bounds__(256) void k_cvt_sign(const float* __restrict__ src,
                                                  _Float16* __restrict__ dst,
                                                  long long n) {
  long long i = ((long long)blockIdx.x * 256 + threadIdx.x) * 4;
  if (i + 3 < n) {
    float4 v = *(const float4*)(src + i);
    v4h o;
    o[0] = (_Float16)(v.x > 0.f ? 1.f : (v.x < 0.f ? -1.f : 0.f));
    o[1] = (_Float16)(v.y > 0.f ? 1.f : (v.y < 0.f ? -1.f : 0.f));
    o[2] = (_Float16)(v.z > 0.f ? 1.f : (v.z < 0.f ? -1.f : 0.f));
    o[3] = (_Float16)(v.w > 0.f ? 1.f : (v.w < 0.f ? -1.f : 0.f));
    *(v4h*)(dst + i) = o;
  }
}

__global__ __launch_bounds__(256) void k_cvt_w2(const float* __restrict__ w2,
                                                _Float16* __restrict__ w2h) {
  long long i = ((long long)blockIdx.x * 256 + threadIdx.x) * 4;
  if (i + 3 < (long long)C_PAD * H_DIM) {
    v4h o;
    if (i < (long long)C_DIM * H_DIM) {
      float4 v = *(const float4*)(w2 + i);
      o[0] = (_Float16)v.x; o[1] = (_Float16)v.y;
      o[2] = (_Float16)v.z; o[3] = (_Float16)v.w;
    } else {
      o[0] = (_Float16)0.f; o[1] = (_Float16)0.f;
      o[2] = (_Float16)0.f; o[3] = (_Float16)0.f;
    }
    *(v4h*)(w2h + i) = o;
  }
}

__global__ __launch_bounds__(256) void k_write_xint(const _Float16* __restrict__ mh,
                                                    const float* __restrict__ sp,
                                                    float* __restrict__ out,
                                                    long long n) {
  long long i = ((long long)blockIdx.x * 256 + threadIdx.x) * 4;
  if (i + 3 < n) {
    const float s = *sp;
    v4h m = *(const v4h*)(mh + i);
    float4 o;
    o.x = s * (float)m[0]; o.y = s * (float)m[1];
    o.z = s * (float)m[2]; o.w = s * (float)m[3];
    *(float4*)(out + i) = o;
  }
}

// ---------------------------------------------------------------------------
extern "C" void kernel_launch(void* const* d_in, const int* in_sizes, int n_in,
                              void* d_out, int out_size, void* d_ws, size_t ws_size,
                              hipStream_t stream) {
  const float* x  = (const float*)d_in[0];  // [B, D]
  const float* W1 = (const float*)d_in[1];  // [H, D]
  const float* b1 = (const float*)d_in[2];  // [H]
  const float* W2 = (const float*)d_in[3];  // [C, H]
  const float* b2 = (const float*)d_in[4];  // [C]

  float* out   = (float*)d_out;                // [B, C]
  float* x_int = out + (size_t)B_DIM * C_DIM;  // [B, H]

  char* ws = (char*)d_ws;
  _Float16* xh   = (_Float16*)(ws);                // 32 MB  x in f16
  _Float16* wsh  = (_Float16*)(ws + 33554432ull);  //  8 MB  sign(W1) f16
  _Float16* w2h  = (_Float16*)(ws + 41943040ull);  //  4 MB  W2 f16 padded
  _Float16* mh   = (_Float16*)(ws + 46137344ull);  // 32 MB  mask f16
  float* part1   = (float*)(ws + 79691776ull);     // 4 KB   gemm1 |hp| partials
  float* part2   = (float*)(ws + 79695872ull);     // 4 KB   |W1| partials
  float* scale_w = (float*)(ws + 79699968ull);
  float* scale_h = (float*)(ws + 79699972ull);

  const long long nW1 = (long long)H_DIM * D_DIM;
  const long long nX  = (long long)B_DIM * D_DIM;
  const long long nBH = (long long)B_DIM * H_DIM;

  // scale_w = mean(|W1|)
  k_reduce_abs<<<1024, 256, 0, stream>>>(W1, nW1, part2);
  k_finalize<<<1, 256, 0, stream>>>(part2, 1024, 1.0f / (float)nW1, scale_w);

  // conversions
  k_cvt_f16 <<<(unsigned)(nX / 4 / 256), 256, 0, stream>>>(x, xh, nX);
  k_cvt_sign<<<(unsigned)(nW1 / 4 / 256), 256, 0, stream>>>(W1, wsh, nW1);
  k_cvt_w2  <<<(unsigned)((long long)C_PAD * H_DIM / 4 / 256), 256, 0, stream>>>(W2, w2h);

  // GEMM1 + mask + |h_pre| partials
  k_gemm1<<<dim3(H_DIM / 128, B_DIM / 128), 256, 0, stream>>>(
      xh, wsh, b1, scale_w, mh, part1);

  // scale_h = mean(|h_pre|)  (16*64 = 1024 partials)
  k_finalize<<<1, 256, 0, stream>>>(part1, 1024, 1.0f / (float)nBH, scale_h);

  // x_internal = scale_h * mask
  k_write_xint<<<(unsigned)(nBH / 4 / 256), 256, 0, stream>>>(mh, scale_h, x_int, nBH);

  // GEMM2 -> out
  k_gemm2<<<dim3(C_PAD / 128, B_DIM / 128), 256, 0, stream>>>(
      mh, w2h, b2, scale_h, out);
}
